// DebertaV2Attention_35261681500706
// MI455X (gfx1250) — compile-verified, matrix operand order checked by reading, not run on hardware
//
#include <hip/hip_runtime.h>
#include <math.h>

// ---------------------------------------------------------------------------
// Problem constants (DeBERTa-v2 attention, B=2 S=1024 DM=1024 H=16 D=64)
// ---------------------------------------------------------------------------
constexpr int kB  = 2;
constexpr int kS  = 1024;
constexpr int kDM = 1024;
constexpr int kH  = 16;
constexpr int kD  = 64;
constexpr int kSpan = 256;          // BUCKETS
constexpr int kP  = 2 * kSpan;      // 512 relative buckets
constexpr int kMaxPos = 512;
#define INV_SCALE 0.07216878364870323f   /* 1/sqrt(64*3) */

typedef __attribute__((ext_vector_type(16))) __bf16 v16bf;
typedef __attribute__((ext_vector_type(8)))  float  v8f;

// ---------------------------------------------------------------------------
// bf16 helpers (workspace storage = unsigned short, RNE rounding)
// ---------------------------------------------------------------------------
__device__ __forceinline__ unsigned short f2bfu(float f) {
  union { float f; unsigned u; } v; v.f = f;
  unsigned r = v.u + 0x7FFFu + ((v.u >> 16) & 1u);
  return (unsigned short)(r >> 16);
}
__device__ __forceinline__ float bfu2f(unsigned short s) {
  union { float f; unsigned u; } v; v.u = ((unsigned)s) << 16; return v.f;
}

// ---------------------------------------------------------------------------
// Fragment loaders. CDNA5 wave32 WMMA layouts:
//  A 16x32 bf16: lane(0-15) row=lane, K {0..7, 16..23}; lane(16-31) row=lane-16,
//  K {8..15, 24..31}.  => per lane exactly two contiguous 16B chunks.
//  B mirrored (lane = column, same K split).
// ---------------------------------------------------------------------------
__device__ __forceinline__ v16bf load_a_frag_g(const unsigned short* __restrict__ A,
                                               int lda, int row0, int k0) {
  int lane = threadIdx.x & 31;
  int rr = lane & 15, half = lane >> 4;
  const unsigned short* p = A + (long)(row0 + rr) * lda + k0 + half * 8;
  union { v16bf v; uint4 q[2]; } u;
  u.q[0] = *(const uint4*)p;          // K base+0..7
  u.q[1] = *(const uint4*)(p + 16);   // K base+16..23
  return u.v;
}

// B 32x16 fragment from row-major N x K storage (multiply by B^T)
__device__ __forceinline__ v16bf load_b_frag_nk(const unsigned short* __restrict__ Bt,
                                                int ldk, int k0, int col0) {
  int lane = threadIdx.x & 31;
  int nn = lane & 15, half = lane >> 4;
  const unsigned short* p = Bt + (long)(col0 + nn) * ldk + k0 + half * 8;
  union { v16bf v; uint4 q[2]; } u;
  u.q[0] = *(const uint4*)p;
  u.q[1] = *(const uint4*)(p + 16);
  return u.v;
}

// ---------------------------------------------------------------------------
// f32 -> bf16 conversion (plain copy)
// ---------------------------------------------------------------------------
__global__ __launch_bounds__(256) void convert_bf16_kernel(
    const float* __restrict__ in, unsigned short* __restrict__ out, long n) {
  long i = (long)blockIdx.x * blockDim.x + threadIdx.x;
  if (i < n) out[i] = f2bfu(in[i]);
}

// ---------------------------------------------------------------------------
// f32 [K][N] -> bf16 [N][K] transpose via LDS tile (weights)
// ---------------------------------------------------------------------------
__global__ __launch_bounds__(256) void transpose_bf16_kernel(
    const float* __restrict__ in, unsigned short* __restrict__ out, int Kd, int Nd) {
  __shared__ float tile[16][17];
  int tx = threadIdx.x & 15, ty = threadIdx.x >> 4;
  tile[ty][tx] = in[(long)(blockIdx.y * 16 + ty) * Nd + blockIdx.x * 16 + tx];
  __syncthreads();
  out[(long)(blockIdx.x * 16 + ty) * Kd + blockIdx.y * 16 + tx] = f2bfu(tile[tx][ty]);
}

// ---------------------------------------------------------------------------
// Static log-bucket relative position gather indices (int16, [S,S])
// ---------------------------------------------------------------------------
__device__ __forceinline__ int log_bucket(int rel) {
  const int mid = kSpan / 2;  // 128
  int sgn = (rel > 0) - (rel < 0);
  int abs_pos = (rel < mid && rel > -mid) ? (mid - 1) : (rel < 0 ? -rel : rel);
  if (abs_pos <= mid) return rel;
  float lp = ceilf(logf((float)abs_pos / (float)mid) /
                   logf((float)(kMaxPos - 1) / (float)mid) * (float)(mid - 1)) + (float)mid;
  return (int)lp * sgn;
}
__device__ __forceinline__ short clamp_idx(int v) {
  return (short)(v < 0 ? 0 : (v > kP - 1 ? kP - 1 : v));
}
__global__ __launch_bounds__(256) void build_idx_kernel(
    short* __restrict__ c2p_idx, short* __restrict__ p2c_idx) {
  int k = blockIdx.x * blockDim.x + threadIdx.x;
  int q = blockIdx.y;
  if (k >= kS) return;
  int bq = log_bucket(q - k);
  int bk = log_bucket(k - q);
  c2p_idx[(long)q * kS + k] = clamp_idx(bq + kSpan);
  p2c_idx[(long)q * kS + k] = clamp_idx(-bk + kSpan);
}

// ---------------------------------------------------------------------------
// Generic WMMA GEMM (bf16 x bf16^T -> f32).
// Block = 128 threads (4 waves); each wave owns TWO 16x16 tiles (32 cols).
// The 16x32 A tile is staged into a double-buffered LDS tile with
// GLOBAL_LOAD_ASYNC_TO_LDS_B64 (all 128 lanes, 8 B each, ASYNCcnt) so tile
// t+1 streams in while tile t feeds the WMMAs.
// Output modes:
//   0: qkv  -> bf16 out[((b*H+h)*S+s)*D+d]           (+bias)
//   1: pos  -> bf16 out[(h*P+row)*D+d]               (+bias)
//   2: c2p  -> bf16 out[z*strideC + row*N + col]
//   3: proj -> f32  out[row*N+col] = acc+bias+resid[row*N+col]
//   4: vT   -> bf16 out[((b*H+h)*D+d)*S+s]           (+bias)
// ---------------------------------------------------------------------------
__global__ __launch_bounds__(128) void wmma_gemm_kernel(
    const unsigned short* __restrict__ A, int lda, long strideA,
    const unsigned short* __restrict__ Bt, int ldk, long strideB, int bmod,
    const float* __restrict__ bias, const float* __restrict__ resid,
    void* __restrict__ out, long strideC,
    int M, int N, int K, int mode) {
  __shared__ __align__(16) unsigned short tileA[2][16][32];  // 2 x 1 KB

  const int tid = threadIdx.x;
  const int w = tid >> 5;
  const int col0 = blockIdx.x * 128 + w * 32;   // wave covers 32 columns
  const int row0 = blockIdx.y * 16;
  const int z = blockIdx.z;

  const unsigned short* Ap = A + (long)z * strideA;
  const unsigned short* Bp = Bt + (long)(bmod > 0 ? (z % bmod) : z) * strideB;

  const unsigned ldsbase = (unsigned)(unsigned long long)(size_t)&tileA[0][0][0];
  const int lane = tid & 31, rr = lane & 15, half = lane >> 4, nn = lane & 15;

  // async staging: all 128 lanes move 8 B of the 16x32 bf16 tile
  const int arow = tid >> 3, ac = tid & 7;
  const unsigned short* gsrc = Ap + (long)(row0 + arow) * lda + ac * 4;
  const unsigned adst = ldsbase + arow * 64 + ac * 8;

  // prologue: stage tile 0 into buffer 0
  asm volatile("global_load_async_to_lds_b64 %0, %1, off"
               :: "v"(adst), "v"(gsrc) : "memory");

  v8f acc0 = {}, acc1 = {};
  for (int k0 = 0; k0 < K; k0 += 32) {
    const int cur = (k0 >> 5) & 1;
    if (k0 + 32 < K) {
      // stage tile t+1 into the other buffer, then wait for tile t only
      asm volatile("global_load_async_to_lds_b64 %0, %1, off"
                   :: "v"(adst + (unsigned)((cur ^ 1) * 1024)),
                      "v"(gsrc + k0 + 32) : "memory");
      asm volatile("s_wait_asynccnt 0x1" ::: "memory");
    } else {
      asm volatile("s_wait_asynccnt 0x0" ::: "memory");
    }
    __syncthreads();

    union { v16bf v; uint4 q[2]; } ua;
    ua.q[0] = *(const uint4*)&tileA[cur][rr][half * 8];
    ua.q[1] = *(const uint4*)&tileA[cur][rr][16 + half * 8];
    v16bf b0 = load_b_frag_nk(Bp, ldk, k0, col0);
    v16bf b1 = load_b_frag_nk(Bp, ldk, k0, col0 + 16);
    if (k0 + 64 < K) {
      __builtin_prefetch(Bp + (long)(col0 + nn) * ldk + k0 + 64, 0, 1);
      __builtin_prefetch(Bp + (long)(col0 + 16 + nn) * ldk + k0 + 64, 0, 1);
    }
    acc0 = __builtin_amdgcn_wmma_f32_16x16x32_bf16(false, ua.v, false, b0,
                                                   (short)0, acc0, false, false);
    acc1 = __builtin_amdgcn_wmma_f32_16x16x32_bf16(false, ua.v, false, b1,
                                                   (short)0, acc1, false, false);
    __syncthreads();
  }

#pragma unroll
  for (int t = 0; t < 2; ++t) {
    v8f acc = t ? acc1 : acc0;
    int col = col0 + t * 16 + nn;
    float bcol = bias ? bias[col] : 0.0f;
#pragma unroll
    for (int r = 0; r < 8; ++r) {
      int row = row0 + r + 8 * half;
      float val = acc[r] + bcol;
      if (mode == 0) {
        int bb = row >> 10, s = row & (kS - 1);
        int hh = col >> 6, d = col & (kD - 1);
        ((unsigned short*)out)[(((long)bb * kH + hh) * kS + s) * kD + d] = f2bfu(val);
      } else if (mode == 1) {
        int hh = col >> 6, d = col & (kD - 1);
        ((unsigned short*)out)[((long)hh * kP + row) * kD + d] = f2bfu(val);
      } else if (mode == 2) {
        ((unsigned short*)out)[(long)z * strideC + (long)row * N + col] = f2bfu(val);
      } else if (mode == 4) {
        int bb = row >> 10, s = row & (kS - 1);
        int hh = col >> 6, d = col & (kD - 1);
        ((unsigned short*)out)[(((long)bb * kH + hh) * kD + d) * kS + s] = f2bfu(val);
      } else {
        ((float*)out)[(long)row * N + col] = val + resid[(long)row * N + col];
      }
    }
  }
}

// ---------------------------------------------------------------------------
// Flash-style attention, 32-wide k blocks.
// One wave per (b, h, 16-row q tile). QK^T: 4 WMMAs / 32 cols; PV: 4 full-K
// WMMAs / 32 cols. V pre-transposed to [B,H,D,S]; probabilities staged in
// LDS as bf16 so the P fragment is two ds_load_b128 per lane.
// ---------------------------------------------------------------------------
__global__ __launch_bounds__(32) void attn_kernel(
    const unsigned short* __restrict__ q,    // [B,H,S,D] bf16
    const unsigned short* __restrict__ k,    // [B,H,S,D] bf16
    const unsigned short* __restrict__ vt,   // [B,H,D,S] bf16
    const unsigned short* __restrict__ c2p,  // [B,H,S,P] bf16
    const unsigned short* __restrict__ p2c,  // [B,H,S,P] bf16
    const short* __restrict__ c2p_idx,       // [S,S]
    const short* __restrict__ p2c_idx,       // [S,S]
    const int* __restrict__ mask,            // [B,1,S,S]
    unsigned short* __restrict__ ctx)        // [B,S,DM] bf16
{
  const int qt = blockIdx.x, h = blockIdx.y, b = blockIdx.z;
  const int lane = threadIdx.x & 31;
  const int half = lane >> 4, nn = lane & 15;
  const long bh = (long)b * kH + h;
  const unsigned short* qs  = q  + bh * (kS * kD);
  const unsigned short* ks  = k  + bh * (kS * kD);
  const unsigned short* vts = vt + bh * (kD * kS);
  const unsigned short* c2ps = c2p + bh * ((long)kS * kP);
  const unsigned short* p2cs = p2c + bh * ((long)kS * kP);

  __shared__ __align__(16) unsigned short probs_bf[16][32];

  float m_r[8], l_r[8];
  v8f o[4] = {};
#pragma unroll
  for (int r = 0; r < 8; ++r) { m_r[r] = -1e30f; l_r[r] = 0.0f; }

  // q A-fragments are loop invariant
  v16bf aq0 = load_a_frag_g(qs, kD, qt * 16, 0);
  v16bf aq1 = load_a_frag_g(qs, kD, qt * 16, 32);

  for (int kt = 0; kt < kS / 32; ++kt) {
    const int c0 = kt * 32;
    // ---- scores for a 16x32 block: 4 WMMAs ----
    v8f s0 = {}, s1 = {};
    {
      v16bf b0a = load_b_frag_nk(ks, kD, 0,  c0);
      v16bf b0b = load_b_frag_nk(ks, kD, 32, c0);
      s0 = __builtin_amdgcn_wmma_f32_16x16x32_bf16(false, aq0, false, b0a, (short)0, s0, false, false);
      s0 = __builtin_amdgcn_wmma_f32_16x16x32_bf16(false, aq1, false, b0b, (short)0, s0, false, false);
      v16bf b1a = load_b_frag_nk(ks, kD, 0,  c0 + 16);
      v16bf b1b = load_b_frag_nk(ks, kD, 32, c0 + 16);
      s1 = __builtin_amdgcn_wmma_f32_16x16x32_bf16(false, aq0, false, b1a, (short)0, s1, false, false);
      s1 = __builtin_amdgcn_wmma_f32_16x16x32_bf16(false, aq1, false, b1b, (short)0, s1, false, false);
    }

    const int colA = c0 + nn, colB = c0 + 16 + nn;
    float sc0[8], sc1[8];
#pragma unroll
    for (int r = 0; r < 8; ++r) {
      int row = qt * 16 + r + 8 * half;
      const short* ci = c2p_idx + (long)row * kS;
      const short* pi = p2c_idx + (long)row * kS;
      float gA = bfu2f(c2ps[(long)row * kP + ci[colA]]) +
                 bfu2f(p2cs[(long)colA * kP + pi[colA]]);
      float gB = bfu2f(c2ps[(long)row * kP + ci[colB]]) +
                 bfu2f(p2cs[(long)colB * kP + pi[colB]]);
      const int* mrow = mask + ((long)b * kS + row) * kS;
      sc0[r] = mrow[colA] ? (s0[r] + gA) * INV_SCALE : -1e30f;
      sc1[r] = mrow[colB] ? (s1[r] + gB) * INV_SCALE : -1e30f;
    }

    // ---- online softmax across 32 columns ----
#pragma unroll
    for (int r = 0; r < 8; ++r) {
      float x = fmaxf(sc0[r], sc1[r]);
      x = fmaxf(x, __shfl_xor(x, 1, 32));
      x = fmaxf(x, __shfl_xor(x, 2, 32));
      x = fmaxf(x, __shfl_xor(x, 4, 32));
      x = fmaxf(x, __shfl_xor(x, 8, 32));
      float newm = fmaxf(m_r[r], x);
      float alpha = __expf(m_r[r] - newm);
      float p0 = __expf(sc0[r] - newm);
      float p1 = __expf(sc1[r] - newm);
      float ps = p0 + p1;
      ps += __shfl_xor(ps, 1, 32);
      ps += __shfl_xor(ps, 2, 32);
      ps += __shfl_xor(ps, 4, 32);
      ps += __shfl_xor(ps, 8, 32);
      l_r[r] = l_r[r] * alpha + ps;
      m_r[r] = newm;
      probs_bf[r + 8 * half][nn]      = f2bfu(p0);
      probs_bf[r + 8 * half][16 + nn] = f2bfu(p1);
#pragma unroll
      for (int dt = 0; dt < 4; ++dt) o[dt][r] *= alpha;
    }
    __syncthreads();

    // ---- PV: P(16x32) x V(32x64): 4 full-K WMMAs ----
    union { v16bf v; uint4 q[2]; } ap;
    {
      int rr = lane & 15;
      ap.q[0] = *(const uint4*)&probs_bf[rr][half * 8];
      ap.q[1] = *(const uint4*)&probs_bf[rr][16 + half * 8];
    }
#pragma unroll
    for (int dt = 0; dt < 4; ++dt) {
      v16bf bv = load_b_frag_nk(vts, kS, c0, dt * 16);
      o[dt] = __builtin_amdgcn_wmma_f32_16x16x32_bf16(false, ap.v, false, bv,
                                                      (short)0, o[dt], false, false);
    }
    __syncthreads();
  }

  // ---- epilogue: normalize and write ctx in [B,S,H*D] ----
#pragma unroll
  for (int r = 0; r < 8; ++r) {
    float inv_l = (l_r[r] > 0.0f) ? (1.0f / l_r[r]) : 0.0f;
    int row = qt * 16 + r + 8 * half;
#pragma unroll
    for (int dt = 0; dt < 4; ++dt) {
      int d = h * kD + dt * 16 + nn;
      ctx[((long)b * kS + row) * kDM + d] = f2bfu(o[dt][r] * inv_l);
    }
  }
}

// ---------------------------------------------------------------------------
// LayerNorm over the residual sum (one block per row)
// ---------------------------------------------------------------------------
__global__ __launch_bounds__(256) void layernorm_kernel(
    const float* __restrict__ h, const float* __restrict__ lnw,
    const float* __restrict__ lnb, float* __restrict__ out) {
  int row = blockIdx.x;
  const float* x = h + (long)row * kDM;
  __shared__ float s1[256], s2[256];
  float vals[4], a = 0.0f, b2 = 0.0f;
#pragma unroll
  for (int i = 0; i < 4; ++i) {
    float v = x[threadIdx.x + 256 * i];
    vals[i] = v; a += v; b2 += v * v;
  }
  s1[threadIdx.x] = a; s2[threadIdx.x] = b2;
  __syncthreads();
  for (int off = 128; off > 0; off >>= 1) {
    if (threadIdx.x < off) {
      s1[threadIdx.x] += s1[threadIdx.x + off];
      s2[threadIdx.x] += s2[threadIdx.x + off];
    }
    __syncthreads();
  }
  float mu = s1[0] * (1.0f / kDM);
  float var = s2[0] * (1.0f / kDM) - mu * mu;
  float rs = rsqrtf(var + 1e-7f);
#pragma unroll
  for (int i = 0; i < 4; ++i) {
    int c = threadIdx.x + 256 * i;
    out[(long)row * kDM + c] = (vals[i] - mu) * rs * lnw[c] + lnb[c];
  }
}

// ---------------------------------------------------------------------------
// Host launcher
// ---------------------------------------------------------------------------
extern "C" void kernel_launch(void* const* d_in, const int* in_sizes, int n_in,
                              void* d_out, int out_size, void* d_ws, size_t ws_size,
                              hipStream_t stream) {
  const float* hidden = (const float*)d_in[0];
  const float* rel    = (const float*)d_in[1];
  const float* Wq = (const float*)d_in[2];  const float* bq = (const float*)d_in[3];
  const float* Wk = (const float*)d_in[4];  const float* bk = (const float*)d_in[5];
  const float* Wv = (const float*)d_in[6];  const float* bv = (const float*)d_in[7];
  const float* Wo = (const float*)d_in[8];  const float* bo = (const float*)d_in[9];
  const float* lnw = (const float*)d_in[10]; const float* lnb = (const float*)d_in[11];
  const int* amask = (const int*)d_in[12];
  (void)in_sizes; (void)n_in; (void)out_size; (void)ws_size;

  char* ws = (char*)d_ws;
  size_t off = 0;
  auto alloc = [&](size_t bytes) -> void* {
    void* p = ws + off;
    off = (off + bytes + 255) & ~(size_t)255;
    return p;
  };

  const long nHid = (long)kB * kS * kDM;
  const long nW   = (long)kDM * kDM;
  const long nRel = (long)kP * kDM;
  const long nQKV = (long)kB * kH * kS * kD;
  const long nPos = (long)kH * kP * kD;
  const long nC2P = (long)kB * kH * kS * kP;

  unsigned short* hid_bf  = (unsigned short*)alloc(nHid * 2);
  unsigned short* Wqt     = (unsigned short*)alloc(nW * 2);    // [N][K] bf16
  unsigned short* Wkt     = (unsigned short*)alloc(nW * 2);
  unsigned short* Wvt     = (unsigned short*)alloc(nW * 2);
  unsigned short* Wot     = (unsigned short*)alloc(nW * 2);
  unsigned short* rel_bf  = (unsigned short*)alloc(nRel * 2);
  unsigned short* q_bf    = (unsigned short*)alloc(nQKV * 2);  // [B,H,S,D]
  unsigned short* k_bf    = (unsigned short*)alloc(nQKV * 2);  // [B,H,S,D]
  unsigned short* vt_bf   = (unsigned short*)alloc(nQKV * 2);  // [B,H,D,S]
  unsigned short* posk_bf = (unsigned short*)alloc(nPos * 2);  // [H,P,D]
  unsigned short* posq_bf = (unsigned short*)alloc(nPos * 2);
  unsigned short* c2p_bf  = (unsigned short*)alloc(nC2P * 2);  // [B,H,S,P]
  unsigned short* p2c_bf  = (unsigned short*)alloc(nC2P * 2);
  unsigned short* ctx_bf  = (unsigned short*)alloc(nHid * 2);  // [B,S,DM]
  float*          h_f     = (float*)alloc(nHid * 4);
  short*          c2p_idx = (short*)alloc((long)kS * kS * 2);
  short*          p2c_idx = (short*)alloc((long)kS * kS * 2);

  auto cv = [&](const float* src, unsigned short* dst, long n) {
    convert_bf16_kernel<<<dim3((unsigned)((n + 255) / 256)), 256, 0, stream>>>(src, dst, n);
  };
  cv(hidden, hid_bf, nHid);
  cv(rel, rel_bf, nRel);

  dim3 gT(kDM / 16, kDM / 16);
  transpose_bf16_kernel<<<gT, 256, 0, stream>>>(Wq, Wqt, kDM, kDM);
  transpose_bf16_kernel<<<gT, 256, 0, stream>>>(Wk, Wkt, kDM, kDM);
  transpose_bf16_kernel<<<gT, 256, 0, stream>>>(Wv, Wvt, kDM, kDM);
  transpose_bf16_kernel<<<gT, 256, 0, stream>>>(Wo, Wot, kDM, kDM);

  build_idx_kernel<<<dim3(kS / 256, kS), 256, 0, stream>>>(c2p_idx, p2c_idx);

  // QKV projections: [2048,1024] x Wt[1024,1024]
  dim3 gQKV(kDM / 128, (kB * kS) / 16, 1);
  wmma_gemm_kernel<<<gQKV, 128, 0, stream>>>(hid_bf, kDM, 0, Wqt, kDM, 0, 0,
      bq, nullptr, q_bf, 0, kB * kS, kDM, kDM, 0);
  wmma_gemm_kernel<<<gQKV, 128, 0, stream>>>(hid_bf, kDM, 0, Wkt, kDM, 0, 0,
      bk, nullptr, k_bf, 0, kB * kS, kDM, kDM, 0);
  wmma_gemm_kernel<<<gQKV, 128, 0, stream>>>(hid_bf, kDM, 0, Wvt, kDM, 0, 0,
      bv, nullptr, vt_bf, 0, kB * kS, kDM, kDM, 4);

  // Positional projections: [512,1024] x Wt -> [H,P,D]
  dim3 gPos(kDM / 128, kP / 16, 1);
  wmma_gemm_kernel<<<gPos, 128, 0, stream>>>(rel_bf, kDM, 0, Wkt, kDM, 0, 0,
      bk, nullptr, posk_bf, 0, kP, kDM, kDM, 1);
  wmma_gemm_kernel<<<gPos, 128, 0, stream>>>(rel_bf, kDM, 0, Wqt, kDM, 0, 0,
      bq, nullptr, posq_bf, 0, kP, kDM, kDM, 1);

  // c2p[b,h] = q[b,h] @ pos_k[h]^T ; p2c[b,h] = k[b,h] @ pos_q[h]^T  (K=64)
  dim3 gBias(kP / 128, kS / 16, kB * kH);
  wmma_gemm_kernel<<<gBias, 128, 0, stream>>>(q_bf, kD, (long)kS * kD,
      posk_bf, kD, (long)kP * kD, kH, nullptr, nullptr,
      c2p_bf, (long)kS * kP, kS, kP, kD, 2);
  wmma_gemm_kernel<<<gBias, 128, 0, stream>>>(k_bf, kD, (long)kS * kD,
      posq_bf, kD, (long)kP * kD, kH, nullptr, nullptr,
      p2c_bf, (long)kS * kP, kS, kP, kD, 2);

  // Attention with online softmax + gathered disentangled bias
  attn_kernel<<<dim3(kS / 16, kH, kB), 32, 0, stream>>>(
      q_bf, k_bf, vt_bf, c2p_bf, p2c_bf, c2p_idx, p2c_idx, amask, ctx_bf);

  // Output projection + residual (f32)
  wmma_gemm_kernel<<<gQKV, 128, 0, stream>>>(ctx_bf, kDM, 0, Wot, kDM, 0, 0,
      bo, hidden, h_f, 0, kB * kS, kDM, kDM, 3);

  // LayerNorm -> d_out
  layernorm_kernel<<<dim3(kB * kS), 256, 0, stream>>>(h_f, lnw, lnb, (float*)d_out);
}
